// TMRModel_8246337208776
// MI455X (gfx1250) — matrix operand order repaired and checked by reading, not compiled, not run on hardware
//
#include <hip/hip_runtime.h>
#include <hip/hip_bf16.h>

typedef __bf16 bf16_t;
typedef __attribute__((ext_vector_type(16))) __bf16 v16bf;
typedef __attribute__((ext_vector_type(8)))  __bf16 v8bf;
typedef __attribute__((ext_vector_type(8)))  float  v8f;

// Problem constants (match reference)
#define BB    32
#define TT    512
#define DD    768
#define SS    64
#define NCLS  8

// ---------------------------------------------------------------------------
// Generic WMMA GEMM: C[b] = A[b] (MxK, row-major bf16) * Bt[b]^T (N x K, row-
// major bf16 => B is KxN), f32 accumulate via v_wmma_f32_16x16x32_bf16.
// One 16x16 output tile per wave32; 8 waves per 256-thread block.
// OUT_MODE: 0 = f32 row-major, 1 = bf16 row-major, 2 = bf16 transposed
// per-batch layout out[(row/tRows)*tBS + col*tLd + row%tRows].
// ---------------------------------------------------------------------------
template<int OUT_MODE>
__global__ __launch_bounds__(256) void wmma_gemm_rr(
    const bf16_t* __restrict__ A, const bf16_t* __restrict__ Bt,
    void* __restrict__ Cv,
    int M, int N, int K,
    long sAb, long sBb, long sCb,
    int lda, int ldb, int ldc,
    int tRows, long tBS, int tLd)
{
  const int wave = blockIdx.x * 8 + (threadIdx.x >> 5);
  const int lane = threadIdx.x & 31;
  const int tilesM = M >> 4, tilesN = N >> 4;
  if (wave >= tilesM * tilesN) return;
  const int tileN = wave / tilesM;
  const int tileM = wave - tileN * tilesM;
  const int batch = blockIdx.y;

  const bf16_t* Ab = A + (long)batch * sAb;
  const bf16_t* Bb = Bt + (long)batch * sBb;

  const int halfSel = lane >> 4;     // lane group 0-15 vs 16-31
  const int sub     = lane & 15;
  const long arow = (long)(tileM * 16 + sub) * lda;
  const long brow = (long)(tileN * 16 + sub) * ldb;
  const int aOff = halfSel * 8;      // A: hi lanes hold K+8..15 / K+24..31
  const int bOff = halfSel * 8;      // B: hi lanes hold K+8..15 / K+24..31

  v8f c = {};
  for (int k0 = 0; k0 < K; k0 += 32) {
    const bf16_t* ap = Ab + arow + k0 + aOff;
    v8bf alo = *(const v8bf*)(ap);
    v8bf ahi = *(const v8bf*)(ap + 16);
    const bf16_t* bp = Bb + brow + k0 + bOff;
    v8bf blo = *(const v8bf*)(bp);
    v8bf bhi = *(const v8bf*)(bp + 16);
    v16bf a, b;
#pragma unroll
    for (int i = 0; i < 8; ++i) {
      a[i] = alo[i]; a[i + 8] = ahi[i];
      b[i] = blo[i]; b[i + 8] = bhi[i];
    }
    c = __builtin_amdgcn_wmma_f32_16x16x32_bf16(
            false, a, false, b, (short)0, c, false, false);
  }

  const int col  = tileN * 16 + sub;
  const int row0 = tileM * 16 + halfSel * 8;
  if (OUT_MODE == 0) {
    float* C = (float*)Cv + (long)batch * sCb;
#pragma unroll
    for (int r = 0; r < 8; ++r) C[(long)(row0 + r) * ldc + col] = c[r];
  } else if (OUT_MODE == 1) {
    bf16_t* C = (bf16_t*)Cv + (long)batch * sCb;
#pragma unroll
    for (int r = 0; r < 8; ++r) C[(long)(row0 + r) * ldc + col] = (bf16_t)c[r];
  } else {
    bf16_t* C = (bf16_t*)Cv;
#pragma unroll
    for (int r = 0; r < 8; ++r) {
      int gr = row0 + r;
      long addr = (long)(gr / tRows) * tBS + (long)col * tLd + (gr % tRows);
      C[addr] = (bf16_t)c[r];
    }
  }
}

// x[b,t,:] = bf16(tok_emb[id] + pos_emb[t])
__global__ __launch_bounds__(256) void embed_bf16(
    const int* __restrict__ ids, const float* __restrict__ tok,
    const float* __restrict__ pos, bf16_t* __restrict__ x)
{
  int row = blockIdx.x;                 // b*T + t
  int t = row & (TT - 1);
  long tb = (long)ids[row] * DD;
  long pb = (long)t * DD;
  long xb = (long)row * DD;
  for (int i = threadIdx.x; i < DD; i += blockDim.x)
    x[xb + i] = (bf16_t)(tok[tb + i] + pos[pb + i]);
}

// WT[e][k] = bf16(W[k][e])
__global__ __launch_bounds__(256) void transpose_to_bf16(
    const float* __restrict__ W, bf16_t* __restrict__ WT, int rows, int cols)
{
  int total = rows * cols;
  for (int i = blockIdx.x * blockDim.x + threadIdx.x; i < total;
       i += gridDim.x * blockDim.x) {
    int k = i / cols, e = i - k * cols;
    WT[(long)e * rows + k] = (bf16_t)W[i];
  }
}

__global__ __launch_bounds__(256) void compute_valid(
    const int* __restrict__ mask, float* __restrict__ valid)
{
  __shared__ float red[256];
  int b = blockIdx.x;
  float s = 0.f;
  for (int t = threadIdx.x; t < TT; t += blockDim.x)
    s += (mask[b * TT + t] > 0) ? 1.f : 0.f;
  red[threadIdx.x] = s; __syncthreads();
  for (int st = 128; st > 0; st >>= 1) {
    if (threadIdx.x < st) red[threadIdx.x] += red[threadIdx.x + st];
    __syncthreads();
  }
  if (threadIdx.x == 0) valid[b] = red[0];
}

__global__ __launch_bounds__(256) void init_mem(
    const float* __restrict__ memi, float* __restrict__ M,
    bf16_t* __restrict__ Mb)
{
  const int total = BB * SS * DD, sd = SS * DD;
  for (int i = blockIdx.x * blockDim.x + threadIdx.x; i < total;
       i += gridDim.x * blockDim.x) {
    float v = memi[i % sd];
    M[i] = v; Mb[i] = (bf16_t)v;
  }
}

// W[b,t,:] = flag * softmax(clip(scores*scale)); stored transposed [b][s][t]
__global__ __launch_bounds__(256) void softmax_rows(
    const float* __restrict__ scores, const int* __restrict__ mask,
    const float* __restrict__ valid, bf16_t* __restrict__ WsmT)
{
  int row = blockIdx.x * 8 + (threadIdx.x >> 5);   // b*T + t
  int lane = threadIdx.x & 31;
  int b = row / TT, t = row - b * TT;
  const float scale = 0.03608439182435161f;        // 1/sqrt(768)
  float s0 = scores[(long)row * SS + lane];
  float s1 = scores[(long)row * SS + lane + 32];
  s0 = fminf(fmaxf(s0 * scale, -20.f), 20.f);
  s1 = fminf(fmaxf(s1 * scale, -20.f), 20.f);
  float mx = fmaxf(s0, s1);
  for (int o = 16; o > 0; o >>= 1) mx = fmaxf(mx, __shfl_xor(mx, o, 32));
  float e0 = __expf(s0 - mx), e1 = __expf(s1 - mx);
  float sm = e0 + e1;
  for (int o = 16; o > 0; o >>= 1) sm += __shfl_xor(sm, o, 32);
  float flag = (mask[row] > 0 || valid[b] == 0.f) ? 1.f : 0.f;
  float inv = flag / sm;
  WsmT[((long)b * SS + lane)      * TT + t] = (bf16_t)(e0 * inv);
  WsmT[((long)b * SS + lane + 32) * TT + t] = (bf16_t)(e1 * inv);
}

// M = clip(0.9*M + 0.1*sigmoid(gate_pre + gate_b)*write, +-50)
__global__ __launch_bounds__(256) void mem_update(
    const float* __restrict__ gp, const float* __restrict__ gbias,
    const float* __restrict__ wr, float* __restrict__ M,
    bf16_t* __restrict__ Mb)
{
  const int total = BB * SS * DD;
  for (int i = blockIdx.x * blockDim.x + threadIdx.x; i < total;
       i += gridDim.x * blockDim.x) {
    float pre = gp[i] + gbias[i % DD];
    float g = 1.f / (1.f + __expf(-pre));
    float v = 0.9f * M[i] + 0.1f * (g * wr[i]);
    v = fminf(fmaxf(v, -50.f), 50.f);
    M[i] = v; Mb[i] = (bf16_t)v;
  }
}

// wbar[b][s] = sum_t mask[b][t] * W[b][t][s]   (reads transposed WsmT)
__global__ __launch_bounds__(256) void wbar_reduce(
    const bf16_t* __restrict__ WsmT, const int* __restrict__ mask,
    float* __restrict__ wbar)
{
  int row = blockIdx.x * 8 + (threadIdx.x >> 5);   // b*S + s, total 2048
  int lane = threadIdx.x & 31;
  int b = row >> 6;
  float acc = 0.f;
  for (int t = lane; t < TT; t += 32)
    acc += (mask[b * TT + t] > 0 ? 1.f : 0.f) * (float)WsmT[(long)row * TT + t];
  for (int o = 16; o > 0; o >>= 1) acc += __shfl_xor(acc, o, 32);
  if (lane == 0) wbar[row] = acc;
}

// pooled[b][d] = (wbar[b] @ M[b]) / max(valid,1)
__global__ __launch_bounds__(256) void pooled_kernel(
    const float* __restrict__ wbar, const float* __restrict__ M,
    const float* __restrict__ valid, float* __restrict__ pooled)
{
  __shared__ float wsh[SS];
  int b = blockIdx.x;
  if (threadIdx.x < SS) wsh[threadIdx.x] = wbar[b * SS + threadIdx.x];
  __syncthreads();
  float inv = 1.f / fmaxf(valid[b], 1.f);
  for (int dd = threadIdx.x; dd < DD; dd += blockDim.x) {
    float acc = 0.f;
    for (int s = 0; s < SS; ++s) acc += wsh[s] * M[((long)b * SS + s) * DD + dd];
    pooled[b * DD + dd] = acc * inv;
  }
}

__global__ __launch_bounds__(256) void ln_logits(
    const float* __restrict__ pooled, const float* __restrict__ lng,
    const float* __restrict__ lnb, const float* __restrict__ clsW,
    const float* __restrict__ clsb, float* __restrict__ out)
{
  __shared__ float red[256];
  int b = blockIdx.x;
  const float* p = pooled + b * DD;
  float s = 0.f;
  for (int dd = threadIdx.x; dd < DD; dd += blockDim.x) s += p[dd];
  red[threadIdx.x] = s; __syncthreads();
  for (int st = 128; st > 0; st >>= 1) {
    if (threadIdx.x < st) red[threadIdx.x] += red[threadIdx.x + st];
    __syncthreads();
  }
  float mu = red[0] / DD; __syncthreads();
  float v = 0.f;
  for (int dd = threadIdx.x; dd < DD; dd += blockDim.x) {
    float t = p[dd] - mu; v += t * t;
  }
  red[threadIdx.x] = v; __syncthreads();
  for (int st = 128; st > 0; st >>= 1) {
    if (threadIdx.x < st) red[threadIdx.x] += red[threadIdx.x + st];
    __syncthreads();
  }
  float rstd = rsqrtf(red[0] / DD + 1e-5f); __syncthreads();
  float acc[NCLS] = {};
  for (int dd = threadIdx.x; dd < DD; dd += blockDim.x) {
    float pn = (p[dd] - mu) * rstd * lng[dd] + lnb[dd];
#pragma unroll
    for (int c = 0; c < NCLS; ++c) acc[c] += pn * clsW[dd * NCLS + c];
  }
  for (int c = 0; c < NCLS; ++c) {
    red[threadIdx.x] = acc[c]; __syncthreads();
    for (int st = 128; st > 0; st >>= 1) {
      if (threadIdx.x < st) red[threadIdx.x] += red[threadIdx.x + st];
      __syncthreads();
    }
    if (threadIdx.x == 0) {
      float r = red[0] + clsb[c];
      if (!isfinite(r)) r = 0.f;
      out[b * NCLS + c] = r;
    }
    __syncthreads();
  }
}

extern "C" void kernel_launch(void* const* d_in, const int* in_sizes, int n_in,
                              void* d_out, int out_size, void* d_ws, size_t ws_size,
                              hipStream_t stream)
{
  (void)in_sizes; (void)n_in; (void)out_size; (void)ws_size;
  const int*   ids   = (const int*)d_in[0];
  const int*   amask = (const int*)d_in[1];
  const float* tok   = (const float*)d_in[2];
  const float* pos   = (const float*)d_in[3];
  const float* Wq    = (const float*)d_in[4];
  const float* Wk    = (const float*)d_in[5];
  const float* Wv    = (const float*)d_in[6];
  const float* gW    = (const float*)d_in[7];
  const float* gb    = (const float*)d_in[8];
  const float* lng   = (const float*)d_in[9];
  const float* lnb   = (const float*)d_in[10];
  const float* clsW  = (const float*)d_in[11];
  const float* clsb  = (const float*)d_in[12];
  const float* memi  = (const float*)d_in[13];
  float* out = (float*)d_out;

  char* base = (char*)d_ws;
  // Region 0 (25,165,824 B): x_bf during projections, then per-step transients
  bf16_t* x_bf   = (bf16_t*)(base + 0);
  bf16_t* Km     = (bf16_t*)(base + 0);           //  3,145,728 B
  float*  scores = (float*) (base + 3145728);     //  4,194,304 B
  bf16_t* WsmT   = (bf16_t*)(base + 7340032);     //  2,097,152 B
  float*  writeb = (float*) (base + 9437184);     //  6,291,456 B
  float*  gateb  = (float*) (base + 15728640);    //  6,291,456 B (ends 22,020,096)
  size_t off = 25165824;
  bf16_t* WqT   = (bf16_t*)(base + off); off += 1179648;
  bf16_t* WkT   = (bf16_t*)(base + off); off += 1179648;
  bf16_t* WvT   = (bf16_t*)(base + off); off += 1179648;
  bf16_t* gWT   = (bf16_t*)(base + off); off += 1179648;
  bf16_t* Qt    = (bf16_t*)(base + off); off += 25165824;
  bf16_t* VtT   = (bf16_t*)(base + off); off += 25165824;   // [b][d][t]
  float*  M_f32 = (float*) (base + off); off += 6291456;
  bf16_t* M_bf  = (bf16_t*)(base + off); off += 3145728;
  float*  valid = (float*) (base + off); off += 256;
  float*  wbar  = (float*) (base + off); off += 8192;
  float*  pooled= (float*) (base + off); off += 98304;

  // ---- setup ----
  embed_bf16<<<BB * TT, 256, 0, stream>>>(ids, tok, pos, x_bf);
  transpose_to_bf16<<<1152, 256, 0, stream>>>(Wq, WqT, DD, DD);
  transpose_to_bf16<<<1152, 256, 0, stream>>>(Wk, WkT, DD, DD);
  transpose_to_bf16<<<1152, 256, 0, stream>>>(Wv, WvT, DD, DD);
  transpose_to_bf16<<<1152, 256, 0, stream>>>(gW, gWT, DD, DD);
  compute_valid<<<BB, 256, 0, stream>>>(amask, valid);
  init_mem<<<1536, 256, 0, stream>>>(memi, M_f32, M_bf);

  // ---- projections: Qt = x@Wq (bf16), VtT = (x@Wv)^T per batch ----
  {
    int tiles = (BB * TT / 16) * (DD / 16);        // 49152
    int blocks = (tiles + 7) / 8;                  // 6144
    wmma_gemm_rr<1><<<dim3(blocks, 1), 256, 0, stream>>>(
        x_bf, WqT, Qt, BB * TT, DD, DD, 0, 0, 0, DD, DD, DD, 0, 0, 0);
    wmma_gemm_rr<2><<<dim3(blocks, 1), 256, 0, stream>>>(
        x_bf, WvT, VtT, BB * TT, DD, DD, 0, 0, 0, DD, DD, 0,
        TT, (long)DD * TT, TT);
  }

  // ---- recurrent steps (4 updates) + final attention read ----
  for (int step = 0; step < 5; ++step) {
    // Km = M @ Wk   (rows = B*S folded)
    wmma_gemm_rr<1><<<dim3(768, 1), 256, 0, stream>>>(
        M_bf, WkT, Km, BB * SS, DD, DD, 0, 0, 0, DD, DD, DD, 0, 0, 0);
    // scores[b] = Qt[b] @ Km[b]^T   (batched, f32 out)
    wmma_gemm_rr<0><<<dim3(16, BB), 256, 0, stream>>>(
        Qt, Km, scores, TT, SS, DD,
        (long)TT * DD, (long)SS * DD, (long)TT * SS, DD, DD, SS, 0, 0, 0);
    // masked/renormalized softmax -> WsmT [b][s][t] bf16
    softmax_rows<<<BB * TT / 8, 256, 0, stream>>>(scores, amask, valid, WsmT);
    if (step == 4) break;
    // write[b] = W^T[b] @ Vt[b]  == WsmT (S x T) * VtT^T  (batched, f32 out)
    wmma_gemm_rr<0><<<dim3(24, BB), 256, 0, stream>>>(
        WsmT, VtT, writeb, SS, DD, TT,
        (long)SS * TT, (long)DD * TT, (long)SS * DD, TT, TT, DD, 0, 0, 0);
    // gate pre-activation = M @ gate_W
    wmma_gemm_rr<0><<<dim3(768, 1), 256, 0, stream>>>(
        M_bf, gWT, gateb, BB * SS, DD, DD, 0, 0, 0, DD, DD, DD, 0, 0, 0);
    // gated decayed update + clip
    mem_update<<<1536, 256, 0, stream>>>(gateb, gb, writeb, M_f32, M_bf);
  }

  // ---- pooled = (m^T W) @ M / max(valid,1); layernorm; logits ----
  wbar_reduce<<<BB * SS / 8, 256, 0, stream>>>(WsmT, amask, wbar);
  pooled_kernel<<<BB, 256, 0, stream>>>(wbar, M_f32, valid, pooled);
  ln_logits<<<BB, 256, 0, stream>>>(pooled, lng, lnb, clsW, clsb, out);
}
